// MultiHeadAttentionStandard_34187939676292
// MI455X (gfx1250) — matrix operand order repaired and checked by reading, not compiled, not run on hardware
//
#include <hip/hip_runtime.h>
#include <hip/hip_bf16.h>

// Problem constants (match reference)
#define Bsz  2
#define Tlen 2048
#define Cdim 1024
#define Hn   16
#define Dd   64
#define Msz  (Bsz * Tlen)   // 4096 rows for the projection GEMMs

typedef __attribute__((ext_vector_type(8)))  _Float16 h8;
typedef __attribute__((ext_vector_type(16))) _Float16 v16h;
typedef __attribute__((ext_vector_type(8)))  float    v8f;

union AFrag { v16h v; h8 h[2]; };

static __device__ __forceinline__ v8f wmma16x16x32(v16h a, v16h b, v8f c) {
  // D(f32 16x16) = A(f16 16x32) * B(f16 32x16) + C
  return __builtin_amdgcn_wmma_f32_16x16x32_f16(
      /*neg_a=*/false, a, /*neg_b=*/false, b,
      /*c_mod=*/(short)0, c, /*reuse_a=*/false, /*reuse_b=*/false);
}

// ---------------------------------------------------------------------------
// fp32 -> fp16 conversion
// ---------------------------------------------------------------------------
__global__ void cvt_f32_f16(const float* __restrict__ src,
                            _Float16* __restrict__ dst, int n) {
  int i = blockIdx.x * blockDim.x + threadIdx.x;
  if (i < n) dst[i] = (_Float16)src[i];
}

// ---------------------------------------------------------------------------
// Y = X @ W^T + bias.  X:[M,C] f16 row-major, W:[C,C] f16 row-major (torch
// nn.Linear weight), bias:[C] f32.
// outMode 0: f16 out at ((b*H+h)*T + t)*D + d      (Q,K head-major)
// outMode 1: f16 out at ((b*H+h)*D + d)*T + t      (V transposed [D,T])
// outMode 2: f32 out at m*C + n                    (final projection)
// One wave computes a 32x64 macro-tile: 2 A-row fragments x 4 B-col fragments
// -> 8 accumulators, 8 wmma per k-step against 12 b128 fragment loads.
// 4 waves per block across N.
// ---------------------------------------------------------------------------
__global__ void gemm_xwT(const _Float16* __restrict__ X,
                         const _Float16* __restrict__ W,
                         const float* __restrict__ bias,
                         _Float16* __restrict__ outH,
                         float* __restrict__ outF,
                         int outMode) {
  const int lane  = threadIdx.x;        // 0..31
  const int g     = lane >> 4;          // lane-half
  const int lm    = lane & 15;
  const int mTile = blockIdx.y * 32;                      // 32 rows per wave
  const int nBase = (blockIdx.x * 4 + threadIdx.y) * 64;  // 64 cols per wave

  const _Float16* xr0 = X + (size_t)(mTile + lm)      * Cdim;  // A rows 0-15
  const _Float16* xr1 = X + (size_t)(mTile + 16 + lm) * Cdim;  // A rows 16-31
  const _Float16* wr[4];
  for (int nt = 0; nt < 4; ++nt)
    wr[nt] = W + (size_t)(nBase + nt * 16 + lm) * Cdim;        // B[k][n]=W[n][k]

  v8f acc[2][4];
  for (int hf = 0; hf < 2; ++hf)
    for (int nt = 0; nt < 4; ++nt) acc[hf][nt] = (v8f){};

  for (int k0 = 0; k0 < Cdim; k0 += 32) {
    if (k0 + 64 < Cdim) {
      __builtin_prefetch(xr0 + k0 + 64, 0, 1);
      __builtin_prefetch(xr1 + k0 + 64, 0, 1);
      __builtin_prefetch(wr[0] + k0 + 64, 0, 1);
    }
    // A (16x32 f16): lanes 0-15 K={0..7,16..23}, lanes 16-31 K={8..15,24..31}
    AFrag a0, a1;
    a0.h[0] = *(const h8*)(xr0 + k0 + g * 8);
    a0.h[1] = *(const h8*)(xr0 + k0 + 16 + g * 8);
    a1.h[0] = *(const h8*)(xr1 + k0 + g * 8);
    a1.h[1] = *(const h8*)(xr1 + k0 + 16 + g * 8);
    // B (32x16 f16): lanes 0-15 K={0..7,16..23}, lanes 16-31 K={8..15,24..31}
    for (int nt = 0; nt < 4; ++nt) {
      AFrag b;
      b.h[0] = *(const h8*)(wr[nt] + k0 + g * 8);
      b.h[1] = *(const h8*)(wr[nt] + k0 + 16 + g * 8);
      acc[0][nt] = wmma16x16x32(a0.v, b.v, acc[0][nt]);
      acc[1][nt] = wmma16x16x32(a1.v, b.v, acc[1][nt]);
    }
  }

  for (int nt = 0; nt < 4; ++nt) {
    const int n  = nBase + nt * 16 + lm;    // C/D: column = lane&15
    const float bn = bias[n];
    const int h = n >> 6, d = n & 63;       // D = 64
    for (int hf = 0; hf < 2; ++hf) {
      for (int r = 0; r < 8; ++r) {
        const int m = mTile + hf * 16 + r + 8 * g;  // C/D: VGPR r -> M=r(+8)
        const float val = acc[hf][nt][r] + bn;
        if (outMode == 2) {
          outF[(size_t)m * Cdim + n] = val;
        } else {
          const int b = m >> 11;            // T = 2048
          const int t = m & (Tlen - 1);
          size_t idx = (outMode == 0)
              ? ((size_t)((b * Hn + h) * Tlen + t) * Dd + d)
              : ((size_t)((b * Hn + h) * Dd + d) * Tlen + t);
          outH[idx] = (_Float16)val;
        }
      }
    }
  }
}

// ---------------------------------------------------------------------------
// Causal flash attention. One wave per (head, 16-row query tile).
// Q,K: [B,H,T,D] f16.  Vt: [B,H,D,T] f16 (transposed).  ctx: [B,T,C] f16.
// Keys processed in chunks of 32; online softmax in f32.
// ---------------------------------------------------------------------------
__global__ void flash_attn(const _Float16* __restrict__ Q,
                           const _Float16* __restrict__ Km,
                           const _Float16* __restrict__ Vt,
                           _Float16* __restrict__ ctx) {
  __shared__ __align__(16) _Float16 pL[16 * 32];  // P tile staging (C/D -> A)

  const int lane = threadIdx.x;
  const int g  = lane >> 4;
  const int lm = lane & 15;
  const int bh = blockIdx.y;            // b*H + h
  const int b  = bh >> 4, h = bh & 15;
  const int qt = blockIdx.x * 16;

  const _Float16* Qh = Q  + (size_t)bh * Tlen * Dd;
  const _Float16* Kh = Km + (size_t)bh * Tlen * Dd;
  const _Float16* Vh = Vt + (size_t)bh * Dd * Tlen;

  // Q A-fragments for the whole tile (reused every chunk): k halves of D=64
  AFrag aq[2];
  for (int f = 0; f < 2; ++f) {
    const _Float16* qr = Qh + (size_t)(qt + lm) * Dd + f * 32;
    aq[f].h[0] = *(const h8*)(qr + g * 16);
    aq[f].h[1] = *(const h8*)(qr + g * 16 + 8);
  }

  v8f o0 = {}, o1 = {}, o2 = {}, o3 = {};      // 16x64 f32 accumulator
  float mrow[8], lrow[8];
  for (int r = 0; r < 8; ++r) { mrow[r] = -3.0e38f; lrow[r] = 0.0f; }

  const int nChunks = (qt + 15) / 32 + 1;      // causal: keys <= qt+15

  for (int kc = 0; kc < nChunks; ++kc) {
    const int kb = kc * 32;

    // ---- S = (Q K^T) for two 16-key tiles ------------------------------
    v8f s0 = {}, s1 = {};
    for (int f = 0; f < 2; ++f) {
      AFrag bk0, bk1;   // B[k][n] = K[kb + n][k]
      const _Float16* k0r = Kh + (size_t)(kb + lm)      * Dd + f * 32 + g * 16;
      const _Float16* k1r = Kh + (size_t)(kb + 16 + lm) * Dd + f * 32 + g * 16;
      bk0.h[0] = *(const h8*)(k0r);  bk0.h[1] = *(const h8*)(k0r + 8);
      bk1.h[0] = *(const h8*)(k1r);  bk1.h[1] = *(const h8*)(k1r + 8);
      s0 = wmma16x16x32(aq[f].v, bk0.v, s0);
      s1 = wmma16x16x32(aq[f].v, bk1.v, s1);
    }

    // ---- scale, causal mask, online softmax ----------------------------
    float p0[8], p1[8];
    for (int r = 0; r < 8; ++r) {
      const int qrow = qt + r + 8 * g;
      float v0 = s0[r] * 0.125f;               // 1/sqrt(64)
      float v1 = s1[r] * 0.125f;
      if (kb + lm      > qrow) v0 = -3.0e38f;
      if (kb + 16 + lm > qrow) v1 = -3.0e38f;
      p0[r] = v0; p1[r] = v1;
    }
    for (int r = 0; r < 8; ++r) {
      float rm = fmaxf(p0[r], p1[r]);
      for (int off = 1; off < 16; off <<= 1)
        rm = fmaxf(rm, __shfl_xor(rm, off, 32));
      const float mnew  = fmaxf(mrow[r], rm);
      const float scale = __expf(mrow[r] - mnew);
      const float e0 = __expf(p0[r] - mnew);
      const float e1 = __expf(p1[r] - mnew);
      float ps = e0 + e1;
      for (int off = 1; off < 16; off <<= 1)
        ps += __shfl_xor(ps, off, 32);
      lrow[r] = lrow[r] * scale + ps;
      mrow[r] = mnew;
      o0[r] *= scale; o1[r] *= scale; o2[r] *= scale; o3[r] *= scale;
      p0[r] = e0; p1[r] = e1;
    }

    // ---- P (C/D layout) -> LDS -> A-fragment layout --------------------
    for (int r = 0; r < 8; ++r) {
      pL[(r + 8 * g) * 32 + lm]      = (_Float16)p0[r];
      pL[(r + 8 * g) * 32 + 16 + lm] = (_Float16)p1[r];
    }
    __syncthreads();
    AFrag ap;
    ap.h[0] = *(const h8*)(&pL[lm * 32 + g * 16]);
    ap.h[1] = *(const h8*)(&pL[lm * 32 + g * 16 + 8]);
    __syncthreads();

    // ---- O += P @ V ; V is [D,T] so B fragments are contiguous ---------
    const _Float16* vb = Vh + kb + g * 16;     // B[k][n] = Vt[n-col][kb+k]
    AFrag bv;
    bv.h[0] = *(const h8*)(vb + (size_t)(lm)      * Tlen);
    bv.h[1] = *(const h8*)(vb + (size_t)(lm)      * Tlen + 8);
    o0 = wmma16x16x32(ap.v, bv.v, o0);
    bv.h[0] = *(const h8*)(vb + (size_t)(16 + lm) * Tlen);
    bv.h[1] = *(const h8*)(vb + (size_t)(16 + lm) * Tlen + 8);
    o1 = wmma16x16x32(ap.v, bv.v, o1);
    bv.h[0] = *(const h8*)(vb + (size_t)(32 + lm) * Tlen);
    bv.h[1] = *(const h8*)(vb + (size_t)(32 + lm) * Tlen + 8);
    o2 = wmma16x16x32(ap.v, bv.v, o2);
    bv.h[0] = *(const h8*)(vb + (size_t)(48 + lm) * Tlen);
    bv.h[1] = *(const h8*)(vb + (size_t)(48 + lm) * Tlen + 8);
    o3 = wmma16x16x32(ap.v, bv.v, o3);
  }

  // ---- normalize and emit context in [B,T,C] layout for the out-proj ----
  for (int r = 0; r < 8; ++r) {
    const int t = qt + r + 8 * g;
    const float inv = 1.0f / lrow[r];
    _Float16* dst = ctx + ((size_t)(b * Tlen + t)) * Cdim + h * Dd + lm;
    dst[0]  = (_Float16)(o0[r] * inv);
    dst[16] = (_Float16)(o1[r] * inv);
    dst[32] = (_Float16)(o2[r] * inv);
    dst[48] = (_Float16)(o3[r] * inv);
  }
}

// ---------------------------------------------------------------------------
// Host-side launch
// ---------------------------------------------------------------------------
extern "C" void kernel_launch(void* const* d_in, const int* in_sizes, int n_in,
                              void* d_out, int out_size, void* d_ws, size_t ws_size,
                              hipStream_t stream) {
  const float* x  = (const float*)d_in[0];
  const float* Wq = (const float*)d_in[1];
  const float* bq = (const float*)d_in[2];
  const float* Wk = (const float*)d_in[3];
  const float* bk = (const float*)d_in[4];
  const float* Wv = (const float*)d_in[5];
  const float* bv = (const float*)d_in[6];
  const float* Wo = (const float*)d_in[7];
  const float* bo = (const float*)d_in[8];
  float* out = (float*)d_out;

  char* ws = (char*)d_ws;
  _Float16* xh  = (_Float16*)(ws + ((size_t)0  << 20));  // 8 MB  x as f16
  _Float16* wqh = (_Float16*)(ws + ((size_t)8  << 20));  // 2 MB
  _Float16* wkh = (_Float16*)(ws + ((size_t)10 << 20));  // 2 MB
  _Float16* wvh = (_Float16*)(ws + ((size_t)12 << 20));  // 2 MB
  _Float16* woh = (_Float16*)(ws + ((size_t)14 << 20));  // 2 MB
  _Float16* Qh  = (_Float16*)(ws + ((size_t)16 << 20));  // 8 MB [B,H,T,D]
  _Float16* Kh  = (_Float16*)(ws + ((size_t)24 << 20));  // 8 MB [B,H,T,D]
  _Float16* Vh  = (_Float16*)(ws + ((size_t)32 << 20));  // 8 MB [B,H,D,T]
  _Float16* ctx = (_Float16*)(ws + ((size_t)40 << 20));  // 8 MB [B,T,C]

  const int nx = Msz * Cdim;        // 4,194,304
  const int nw = Cdim * Cdim;       // 1,048,576
  cvt_f32_f16<<<(nx + 255) / 256, 256, 0, stream>>>(x,  xh,  nx);
  cvt_f32_f16<<<(nw + 255) / 256, 256, 0, stream>>>(Wq, wqh, nw);
  cvt_f32_f16<<<(nw + 255) / 256, 256, 0, stream>>>(Wk, wkh, nw);
  cvt_f32_f16<<<(nw + 255) / 256, 256, 0, stream>>>(Wv, wvh, nw);
  cvt_f32_f16<<<(nw + 255) / 256, 256, 0, stream>>>(Wo, woh, nw);

  dim3 gGrid(Cdim / 256, Msz / 32); // (4, 128) blocks
  dim3 gBlk(32, 4);                 // 4 waves, each a 32x64 macro-tile
  gemm_xwT<<<gGrid, gBlk, 0, stream>>>(xh, wqh, bq, Qh, nullptr, 0);
  gemm_xwT<<<gGrid, gBlk, 0, stream>>>(xh, wkh, bk, Kh, nullptr, 0);
  gemm_xwT<<<gGrid, gBlk, 0, stream>>>(xh, wvh, bv, Vh, nullptr, 1);

  dim3 aGrid(Tlen / 16, Bsz * Hn);  // (128, 32) waves
  flash_attn<<<aGrid, 32, 0, stream>>>(Qh, Kh, Vh, ctx);

  gemm_xwT<<<gGrid, gBlk, 0, stream>>>(ctx, woh, bo, nullptr, out, 2);
}
